// ParallelLSTM_77489799954528
// MI455X (gfx1250) — matrix operand order repaired and checked by reading, not compiled, not run on hardware
//
#include <hip/hip_runtime.h>

// Problem constants (B=64, T=196, D=768, H=12, HID=64)
#define B_    64
#define T_    196
#define D_    768
#define H_    12
#define HID_  64
#define BT_   (B_*T_)        // 12544 = 98*128
#define GATES_ 256           // 4*HID
#define KC_   128            // fused K: [x_t | h]
#define DH_   24             // 2 dirs * 12 heads

// padded LDS row strides (u16 units); multiples of 8 (16B) and co-prime-ish with 64 banks
#define LDA_SM 40            // gemm A/B tile row stride (32 cols + pad)
#define XT_LD  72            // x-tile / h row stride (64 cols + pad)

typedef unsigned short u16;
typedef __attribute__((ext_vector_type(8)))  unsigned short us8;   // 16B
typedef __attribute__((ext_vector_type(16))) __bf16 v16bf;
typedef __attribute__((ext_vector_type(8)))  float  v8f;
typedef __attribute__((ext_vector_type(4)))  int    v4i;           // 16B

union FragU { v16bf v; us8 h[2]; };

#define WMMA_BF16(a, b, c) \
  __builtin_amdgcn_wmma_f32_16x16x32_bf16(false, (a), false, (b), (short)0, (c), false, false)

// ---- CDNA5 async global->LDS path (guarded; sync fallback keeps it compiling)
#if defined(__gfx1250__) && __has_builtin(__builtin_amdgcn_global_load_async_to_lds_b128)
#define HAVE_ASYNC 1
#else
#define HAVE_ASYNC 0
#endif

__device__ __forceinline__ void async_cp128(const void* g, void* l) {
#if HAVE_ASYNC
  // signature (from hipcc diagnostic): (v4i AS1*, v4i AS3*, imm offset, imm cpol)
  __builtin_amdgcn_global_load_async_to_lds_b128(
      (__attribute__((address_space(1))) v4i*)g,
      (__attribute__((address_space(3))) v4i*)l, 0, 0);
#else
  *(us8*)l = *(const us8*)g;
#endif
}

#if HAVE_ASYNC
#if __has_builtin(__builtin_amdgcn_s_wait_asynccnt)
#define ASYNC_WAIT(n) __builtin_amdgcn_s_wait_asynccnt(n)
#else
#define ASYNC_WAIT(n) asm volatile("s_wait_asynccnt %0" ::"i"(n) : "memory")
#endif
#else
#define ASYNC_WAIT(n) ((void)0)
#endif

__device__ __forceinline__ u16 f2b(float f) {
  unsigned u = __float_as_uint(f);
  return (u16)((u + 0x7FFFu + ((u >> 16) & 1u)) >> 16);  // RNE
}
__device__ __forceinline__ float b2f(u16 b) {
  return __uint_as_float(((unsigned)b) << 16);
}
__device__ __forceinline__ float fast_tanh(float x) {
#if __has_builtin(__builtin_amdgcn_tanhf)
  return __builtin_amdgcn_tanhf(x);   // v_tanh_f32 (TRANS)
#else
  return tanhf(x);
#endif
}
__device__ __forceinline__ float sigf(float x) {   // sigmoid via one TRANS op
  return __builtin_fmaf(0.5f, fast_tanh(0.5f * x), 0.5f);
}

// A fragment: 16x32 bf16 tile from row-major [*, ld] source.
// lane<16: M=lane, K 0-7 (h[0]) and 16-23 (h[1]); lane>=16: M=lane-16, K 8-15 / 24-31.
__device__ __forceinline__ v16bf load_a(const u16* base, int ld, int m0, int k0, int lane) {
  const u16* p = base + (long)(m0 + (lane & 15)) * ld + k0 + ((lane & 16) >> 1);
  FragU f;
  f.h[0] = *(const us8*)p;
  f.h[1] = *(const us8*)(p + 16);
  return f.v;
}

// B fragment: 32x16 bf16 tile (KxN) from weights stored [N,K] row-major (y = x @ W^T).
__device__ __forceinline__ v16bf load_b(const u16* W, int ld, int n0, int k0, int lane) {
  const u16* p = W + (long)(n0 + (lane & 15)) * ld + k0 + ((lane & 16) >> 1);
  FragU f;
  f.h[0] = *(const us8*)p;
  f.h[1] = *(const us8*)(p + 16);
  return f.v;
}

// ---------------------------------------------------------------------------
__global__ void __launch_bounds__(256) cvt_bf16(const float* __restrict__ in,
                                                u16* __restrict__ out, int n) {
  int i = blockIdx.x * 256 + threadIdx.x;
  if (i < n) out[i] = f2b(in[i]);
}

// Fused recurrent weights Wc[dir,head][n][0..63]=Wih, [64..127]=Whh (bf16); bsum=bih+bhh.
__global__ void __launch_bounds__(256) pack_comb(
    const float* __restrict__ Wih_f, const float* __restrict__ Whh_f,
    const float* __restrict__ bih_f, const float* __restrict__ bhh_f,
    const float* __restrict__ Wih_r, const float* __restrict__ Whh_r,
    const float* __restrict__ bih_r, const float* __restrict__ bhh_r,
    u16* __restrict__ Wc, float* __restrict__ bsum) {
  int idx = blockIdx.x * 256 + threadIdx.x;
  if (idx >= DH_ * GATES_ * KC_) return;
  int k  = idx & (KC_ - 1);
  int nn = (idx >> 7) & (GATES_ - 1);
  int dh = idx >> 15;                 // 0..23
  int dir = dh / H_, head = dh % H_;
  const float* Wih = dir ? Wih_r : Wih_f;
  const float* Whh = dir ? Whh_r : Whh_f;
  float v = (k < HID_) ? Wih[((size_t)head * GATES_ + nn) * HID_ + k]
                       : Whh[((size_t)head * GATES_ + nn) * HID_ + (k - HID_)];
  Wc[idx] = f2b(v);
  if (k == 0) {
    const float* bi = dir ? bih_r : bih_f;
    const float* bh = dir ? bhh_r : bhh_f;
    bsum[dh * GATES_ + nn] = bi[head * GATES_ + nn] + bh[head * GATES_ + nn];
  }
}

// ---------------------------------------------------------------------------
// WMMA GEMM: out[M,N] = A[M,K](bf16) @ W[N,K]^T(bf16) + bias[N]
// 128x128 block tile; K staged 32 at a time into LDS with async double buffering.
__global__ void __launch_bounds__(256) wmma_gemm_bias(
    const u16* __restrict__ A, const u16* __restrict__ W,
    const float* __restrict__ bias, u16* __restrict__ outb,
    float* __restrict__ outf, int M, int N, int K) {
  __shared__ u16 Asm[2][128 * LDA_SM];   // 2 x 10 KB
  __shared__ u16 Bsm[2][128 * LDA_SM];   // 2 x 10 KB

  const int tid  = threadIdx.x;
  const int lane = tid & 31;
  const int wave = tid >> 5;
  const int bm = blockIdx.y * 128;
  const int bn = blockIdx.x * 128;
  const int mtl = (wave >> 1) * 32;      // wave-local m-tiles: mtl, mtl+16
  const int ntl = (wave & 1) * 64;       // wave-local n-tiles: ntl + {0,16,32,48}

  // stage k-slice [k0, k0+32) of A and W tiles into LDS buffer `buf`
  auto issue_stage = [&](int k0, int buf) {
#pragma unroll
    for (int j = 0; j < 2; ++j) {
      int idx = tid + j * 256;           // 0..511
      int row = idx >> 2;                // 0..127
      int seg = idx & 3;                 // 4 x 8 elements = 32 cols
      async_cp128(A + (size_t)(bm + row) * K + k0 + seg * 8,
                  &Asm[buf][row * LDA_SM + seg * 8]);
      async_cp128(W + (size_t)(bn + row) * K + k0 + seg * 8,
                  &Bsm[buf][row * LDA_SM + seg * 8]);
    }
  };

  v8f acc[2][4];
#pragma unroll
  for (int j = 0; j < 4; ++j) {
    float bv = bias[bn + ntl + j * 16 + (lane & 15)];
    v8f c = {bv, bv, bv, bv, bv, bv, bv, bv};
    acc[0][j] = c;
    acc[1][j] = c;
  }

  issue_stage(0, 0);
  int buf = 0;
  for (int k0 = 0; k0 < K; k0 += 32) {
    if (k0 + 32 < K) {
      issue_stage(k0 + 32, buf ^ 1);     // overlap next stage with this one's math
      ASYNC_WAIT(4);                     // only the 4 just-issued may remain outstanding
    } else {
      ASYNC_WAIT(0);
    }
    __syncthreads();                     // all waves' stage-k data visible in LDS

    v16bf a0 = load_a(Asm[buf], LDA_SM, mtl, 0, lane);
    v16bf a1 = load_a(Asm[buf], LDA_SM, mtl + 16, 0, lane);
#pragma unroll
    for (int j = 0; j < 4; ++j) {
      v16bf b = load_b(Bsm[buf], LDA_SM, ntl + j * 16, 0, lane);
      acc[0][j] = WMMA_BF16(a0, b, acc[0][j]);
      acc[1][j] = WMMA_BF16(a1, b, acc[1][j]);
    }
    __syncthreads();                     // safe to overwrite this buffer next-next stage
    buf ^= 1;
  }

  // C/D layout: VGPR i holds M = m0 + i + (lane>=16 ? 8 : 0), N = n0 + lane%16
#pragma unroll
  for (int i2 = 0; i2 < 2; ++i2)
#pragma unroll
    for (int j = 0; j < 4; ++j) {
      int n = bn + ntl + j * 16 + (lane & 15);
#pragma unroll
      for (int i = 0; i < 8; ++i) {
        long m = bm + mtl + i2 * 16 + i + ((lane & 16) >> 1);
        float v = acc[i2][j][i];
        if (outf) outf[m * N + n] = v;
        else      outb[m * N + n] = f2b(v);
      }
    }
}

// ---------------------------------------------------------------------------
// Bidirectional LSTM recurrence. One block per (dir, head); 256 threads = 8 waves.
// Per step: gates[64,256] = [x_t | h](64x128) @ Wc^T + b via WMMA (Wc in regs),
// x_t tile prefetched one step ahead (async -> LDS), h in LDS, c in VGPRs.
__global__ void __launch_bounds__(256) bilstm_rec(
    const u16* __restrict__ xpre,   // [B*T, D] bf16, row m = b*T + t
    const u16* __restrict__ Wc,     // [24, 256, 128] bf16
    const float* __restrict__ bsum, // [24, 256]
    u16* __restrict__ ycat) {       // [B, T, 2*H*HID] bf16
  __shared__ u16 xt_lds[2][B_ * XT_LD];  // 2 x 9 KB  (x_t tile, padded rows)
  __shared__ u16 h_lds[B_ * XT_LD];      // 9 KB      (h, padded rows)
  __shared__ u16 g_lds[B_ * GATES_];     // 32 KB     (activated gates, bf16)

  const int tid  = threadIdx.x;
  const int lane = tid & 31;
  const int wave = tid >> 5;
  const int dh   = blockIdx.x;     // 0..23
  const int dir  = dh / H_;
  const int head = dh % H_;

  const u16*   Wci = Wc + (size_t)dh * GATES_ * KC_;
  const float* bs  = bsum + dh * GATES_;
  const int n0w = wave * 32;       // this wave owns gate columns [n0w, n0w+32)

  // Loop-invariant B fragments: 2 n-tiles x 4 k-steps, kept in VGPRs.
  v16bf bf[2][4];
  float bv[2];
#pragma unroll
  for (int j = 0; j < 2; ++j) {
    bv[j] = bs[n0w + j * 16 + (lane & 15)];
#pragma unroll
    for (int k = 0; k < 4; ++k)
      bf[j][k] = load_b(Wci, KC_, n0w + j * 16, k * 32, lane);
  }

  // async prefetch of the x_t tile (64 rows x 64 cols bf16) for timestep t
  auto issue_xtile = [&](int t, int buf) {
#pragma unroll
    for (int j = 0; j < 2; ++j) {
      int idx = tid + j * 256;          // 0..511
      int row = idx >> 3;               // batch b: 0..63
      int seg = idx & 7;                // 8 x 8 elements = 64 cols
      async_cp128(xpre + ((size_t)row * T_ + t) * D_ + head * HID_ + seg * 8,
                  &xt_lds[buf][row * XT_LD + seg * 8]);
    }
  };

  for (int i = tid; i < B_ * XT_LD; i += 256) h_lds[i] = 0;
  float creg[16];
#pragma unroll
  for (int q = 0; q < 16; ++q) creg[q] = 0.0f;

  const int um  = tid >> 2;        // batch row for update phase
  const int uj0 = (tid & 3) << 4;  // 16 hidden cols

  issue_xtile(dir ? T_ - 1 : 0, 0);
  int xbuf = 0;

  for (int s = 0; s < T_; ++s) {
    const int t = dir ? (T_ - 1 - s) : s;
    if (s + 1 < T_) {
      issue_xtile(dir ? T_ - 2 - s : s + 1, xbuf ^ 1);
      ASYNC_WAIT(2);                    // tile for step s has landed
    } else {
      ASYNC_WAIT(0);
    }
    __syncthreads();  // x tile + prev-step h visible to all waves

    for (int mt = 0; mt < 4; ++mt) {
      const int m0 = mt * 16;
      v16bf ax0 = load_a(xt_lds[xbuf], XT_LD, m0, 0,  lane);
      v16bf ax1 = load_a(xt_lds[xbuf], XT_LD, m0, 32, lane);
      v16bf ah0 = load_a(h_lds,        XT_LD, m0, 0,  lane);
      v16bf ah1 = load_a(h_lds,        XT_LD, m0, 32, lane);
#pragma unroll
      for (int j = 0; j < 2; ++j) {
        v8f c = {bv[j], bv[j], bv[j], bv[j], bv[j], bv[j], bv[j], bv[j]};
        c = WMMA_BF16(ax0, bf[j][0], c);
        c = WMMA_BF16(ax1, bf[j][1], c);
        c = WMMA_BF16(ah0, bf[j][2], c);
        c = WMMA_BF16(ah1, bf[j][3], c);
        const int gate = (n0w + j * 16) >> 6;       // 0:i 1:f 2:g 3:o (uniform per tile)
        const int ncol = n0w + j * 16 + (lane & 15);
#pragma unroll
        for (int i = 0; i < 8; ++i) {
          float v = c[i];
          v = (gate == 2) ? fast_tanh(v) : sigf(v);
          g_lds[(m0 + i + ((lane & 16) >> 1)) * GATES_ + ncol] = f2b(v);
        }
      }
    }
    __syncthreads();  // all gates in LDS; x buffer xbuf free for re-fill

    // Cell update: thread owns (um, uj0..uj0+15); c stays in registers.
    alignas(16) u16 hout[16];
    const u16* gr = g_lds + um * GATES_;
#pragma unroll
    for (int q = 0; q < 16; ++q) {
      const int j = uj0 + q;
      float gi = b2f(gr[j]);
      float gf = b2f(gr[64 + j]);
      float gg = b2f(gr[128 + j]);
      float go = b2f(gr[192 + j]);
      float cn = gf * creg[q] + gi * gg;
      creg[q] = cn;
      u16 hb = f2b(go * fast_tanh(cn));
      hout[q] = hb;
      h_lds[um * XT_LD + j] = hb;
    }
    u16* yp = ycat + ((size_t)um * T_ + t) * (2 * H_ * HID_) + dir * (H_ * HID_) +
              head * HID_ + uj0;
    *(us8*)yp = *(const us8*)hout;
    *(us8*)(yp + 8) = *(const us8*)(hout + 8);

    xbuf ^= 1;
  }
}

// ---------------------------------------------------------------------------
extern "C" void kernel_launch(void* const* d_in, const int* in_sizes, int n_in,
                              void* d_out, int out_size, void* d_ws, size_t ws_size,
                              hipStream_t stream) {
  const float* x      = (const float*)d_in[0];
  const float* pre_W  = (const float*)d_in[1];
  const float* pre_b  = (const float*)d_in[2];
  const float* Wih_f  = (const float*)d_in[3];
  const float* Whh_f  = (const float*)d_in[4];
  const float* bih_f  = (const float*)d_in[5];
  const float* bhh_f  = (const float*)d_in[6];
  const float* Wih_r  = (const float*)d_in[7];
  const float* Whh_r  = (const float*)d_in[8];
  const float* bih_r  = (const float*)d_in[9];
  const float* bhh_r  = (const float*)d_in[10];
  const float* proj_W = (const float*)d_in[11];
  const float* proj_b = (const float*)d_in[12];
  float* out = (float*)d_out;
  (void)in_sizes; (void)n_in; (void)out_size; (void)ws_size;

  char* w = (char*)d_ws;
  size_t off = 0;
  auto alloc = [&](size_t bytes) -> char* {
    char* p = w + off;
    off = (off + bytes + 255) & ~(size_t)255;
    return p;
  };
  u16*   xb     = (u16*)alloc((size_t)BT_ * D_ * 2);            // x in bf16
  u16*   preWb  = (u16*)alloc((size_t)D_ * D_ * 2);
  u16*   projWb = (u16*)alloc((size_t)D_ * 2 * H_ * HID_ * 2);
  u16*   Wc     = (u16*)alloc((size_t)DH_ * GATES_ * KC_ * 2);
  float* bsum   = (float*)alloc((size_t)DH_ * GATES_ * 4);
  u16*   xpre   = (u16*)alloc((size_t)BT_ * D_ * 2);            // pre-proj out (bf16)
  u16*   ycat   = (u16*)alloc((size_t)BT_ * 2 * H_ * HID_ * 2); // concat LSTM out

  int n;
  n = BT_ * D_;
  cvt_bf16<<<(n + 255) / 256, 256, 0, stream>>>(x, xb, n);
  n = D_ * D_;
  cvt_bf16<<<(n + 255) / 256, 256, 0, stream>>>(pre_W, preWb, n);
  n = D_ * 2 * H_ * HID_;
  cvt_bf16<<<(n + 255) / 256, 256, 0, stream>>>(proj_W, projWb, n);
  n = DH_ * GATES_ * KC_;
  pack_comb<<<(n + 255) / 256, 256, 0, stream>>>(Wih_f, Whh_f, bih_f, bhh_f,
                                                 Wih_r, Whh_r, bih_r, bhh_r, Wc, bsum);

  // Phase A: xpre = x @ pre_W^T + pre_b   (12544 x 768 x 768), bf16 out
  wmma_gemm_bias<<<dim3(D_ / 128, BT_ / 128), 256, 0, stream>>>(
      xb, preWb, pre_b, xpre, nullptr, BT_, D_, D_);

  // Phase B+C: fused gate GEMM + recurrence per (dir, head)
  bilstm_rec<<<DH_, 256, 0, stream>>>(xpre, Wc, bsum, ycat);

  // Phase D: out = ycat @ proj_W^T + proj_b   (12544 x 768 x 1536), f32 out
  wmma_gemm_bias<<<dim3(D_ / 128, BT_ / 128), 256, 0, stream>>>(
      ycat, projWb, proj_b, nullptr, out, BT_, D_, 2 * H_ * HID_);
}
